// DynamicTemporalModel_39530878992637
// MI455X (gfx1250) — matrix operand order repaired and checked by reading, not compiled, not run on hardware
//
#include <hip/hip_runtime.h>

// ---------------------------------------------------------------------------
// DynamicTemporalModel for MI455X (gfx1250), wave32 + WMMA bf16 (f32 accum).
// ---------------------------------------------------------------------------

typedef __attribute__((ext_vector_type(16))) __bf16 v16bf;
typedef __attribute__((ext_vector_type(8)))  float  v8f;

#define kB   64
#define kT   128
#define kD   1024
#define kH   8
#define kHD  128
#define kC   1000
#define kS   5
#define kM   10
#define kBT  (kB * kT)

__device__ __forceinline__ v8f zero8() {
  v8f z = {0.f, 0.f, 0.f, 0.f, 0.f, 0.f, 0.f, 0.f};
  return z;
}

__device__ __forceinline__ float sigf(float x) { return 1.f / (1.f + __expf(-x)); }

__device__ __forceinline__ v8f wmma_bf16(v16bf a, v16bf b, v8f c) {
  return __builtin_amdgcn_wmma_f32_16x16x32_bf16(false, a, false, b, (short)0, c,
                                                 false, false);
}

// A-fragment: 16x32 bf16 tile, rows (m0..m0+15) of `base` (row stride = stride).
// Contiguous-k layout used here (not the ISA table order): lane L holds row M=L%15? -> M=L&15, half=L>>4;
// VGPR j: K = 16*half + 2*j + {0,1}
__device__ __forceinline__ v16bf load_a_frag(const __bf16* base, int stride,
                                             int m0, int lane) {
  int Mr = lane & 15, half = lane >> 4;
  const __bf16* row = base + (m0 + Mr) * stride;
  v16bf a;
#pragma unroll
  for (int j = 0; j < 8; ++j) {
    int k = 16 * half + 2 * j;
    a[2 * j]     = row[k];
    a[2 * j + 1] = row[k + 1];
  }
  return a;
}

// B-fragment: B[k][n] = base[(n0+n)*stride + k]  (i.e. W stored row-major [n][k]).
// lane L holds col N=L&15, half=L>>4; VGPR j: K = 16*half + 2*j + {0,1}
__device__ __forceinline__ v16bf load_b_frag(const __bf16* base, int stride,
                                             int n0, int lane) {
  int Nc = lane & 15, half = lane >> 4;
  const __bf16* row = base + (n0 + Nc) * stride;
  v16bf b;
#pragma unroll
  for (int j = 0; j < 8; ++j) {
    int k = 16 * half + 2 * j;
    b[2 * j]     = row[k];
    b[2 * j + 1] = row[k + 1];
  }
  return b;
}

// ---------------------------------------------------------------------------
// Generic GEMM: Y[N,Mo] = X[N,K] @ W[Mo,K]^T + bias1 + bias2, optional act.
// 64x64 tile per block, 4 waves, each wave one 32x32 quadrant (2x2 WMMA).
// act: 0=none, 1=relu, 2=sigmoid
// ---------------------------------------------------------------------------
__global__ __launch_bounds__(128) void gemm_bias_act(
    const float* __restrict__ X, const float* __restrict__ W,
    const float* __restrict__ bias1, const float* __restrict__ bias2,
    float* __restrict__ Y, int N, int K, int Mo, int act) {
  __shared__ __bf16 Xs[64 * 40];
  __shared__ __bf16 Ws[64 * 40];
  int bn = blockIdx.x * 64;
  int bm = blockIdx.y * 64;
  int tid = threadIdx.x, lane = tid & 31, wv = tid >> 5;
  int rw = (wv & 1) * 32, cw = (wv >> 1) * 32;

  v8f acc[2][2];
#pragma unroll
  for (int i = 0; i < 2; ++i)
#pragma unroll
    for (int j = 0; j < 2; ++j) acc[i][j] = zero8();

  for (int k0 = 0; k0 < K; k0 += 32) {
    for (int i = tid; i < 64 * 32; i += 128) {
      int r = i >> 5, c = i & 31;
      Xs[r * 40 + c] = (__bf16)X[(size_t)(bn + r) * K + k0 + c];
      Ws[r * 40 + c] = (__bf16)W[(size_t)(bm + r) * K + k0 + c];
    }
    if (k0 + 32 < K) {  // hint next X tile toward the caches
      __builtin_prefetch(X + (size_t)(bn + (tid & 63)) * K + k0 + 32, 0, 0);
    }
    __syncthreads();
    v16bf a0 = load_a_frag(Xs, 40, rw, lane);
    v16bf a1 = load_a_frag(Xs, 40, rw + 16, lane);
    v16bf b0 = load_b_frag(Ws, 40, cw, lane);
    v16bf b1 = load_b_frag(Ws, 40, cw + 16, lane);
    acc[0][0] = wmma_bf16(a0, b0, acc[0][0]);
    acc[0][1] = wmma_bf16(a0, b1, acc[0][1]);
    acc[1][0] = wmma_bf16(a1, b0, acc[1][0]);
    acc[1][1] = wmma_bf16(a1, b1, acc[1][1]);
    __syncthreads();
  }

  int half = lane >> 4, c16 = lane & 15;
#pragma unroll
  for (int ti = 0; ti < 2; ++ti)
#pragma unroll
    for (int tj = 0; tj < 2; ++tj)
#pragma unroll
      for (int r = 0; r < 8; ++r) {
        int row = bn + rw + ti * 16 + r + 8 * half;
        int col = bm + cw + tj * 16 + c16;
        float v = acc[ti][tj][r];
        if (bias1) v += bias1[col];
        if (bias2) v += bias2[col];
        if (act == 1)
          v = fmaxf(v, 0.f);
        else if (act == 2)
          v = sigf(v);
        Y[(size_t)row * Mo + col] = v;
      }
}

// ---------------------------------------------------------------------------
// Stage encoder: Linear(2,64)->LN->ReLU->Linear(64,1024)->LN, + x  -> xp
// one block per token.
// ---------------------------------------------------------------------------
__global__ __launch_bounds__(256) void encode_add(
    const float* __restrict__ x, const int* __restrict__ stage,
    const float* __restrict__ w1, const float* __restrict__ b1,
    const float* __restrict__ g1, const float* __restrict__ bb1,
    const float* __restrict__ w2, const float* __restrict__ b2,
    const float* __restrict__ g2, const float* __restrict__ bb2,
    float* __restrict__ xp) {
  __shared__ float h1[64];
  __shared__ float red[256], red2[256];
  __shared__ float stats[2];
  int tok = blockIdx.x;
  int t = tok & (kT - 1);
  int tid = threadIdx.x;
  float pos = (float)t, stg = (float)stage[tok];
  if (tid < 64) h1[tid] = w1[tid * 2] * pos + w1[tid * 2 + 1] * stg + b1[tid];
  __syncthreads();
  if (tid < 32) {
    float v = h1[tid] + h1[tid + 32];
    float v2 = h1[tid] * h1[tid] + h1[tid + 32] * h1[tid + 32];
#pragma unroll
    for (int m = 16; m >= 1; m >>= 1) {
      v += __shfl_xor(v, m, 32);
      v2 += __shfl_xor(v2, m, 32);
    }
    if (tid == 0) {
      float mu = v / 64.f;
      float var = v2 / 64.f - mu * mu;
      stats[0] = mu;
      stats[1] = rsqrtf(var + 1e-5f);
    }
  }
  __syncthreads();
  if (tid < 64)
    h1[tid] = fmaxf((h1[tid] - stats[0]) * stats[1] * g1[tid] + bb1[tid], 0.f);
  __syncthreads();

  float e[4];
  float s = 0.f, s2 = 0.f;
#pragma unroll
  for (int q = 0; q < 4; ++q) {
    int j = tid * 4 + q;
    float acc = b2[j];
    const float* wr = w2 + (size_t)j * 64;
    for (int k = 0; k < 64; ++k) acc += h1[k] * wr[k];
    e[q] = acc;
    s += acc;
    s2 += acc * acc;
  }
  red[tid] = s;
  red2[tid] = s2;
  __syncthreads();
  for (int off = 128; off >= 1; off >>= 1) {
    if (tid < off) {
      red[tid] += red[tid + off];
      red2[tid] += red2[tid + off];
    }
    __syncthreads();
  }
  float mu = red[0] / 1024.f;
  float var = red2[0] / 1024.f - mu * mu;
  float rstd = rsqrtf(var + 1e-5f);
  const float* xr = x + (size_t)tok * kD;
  float* xo = xp + (size_t)tok * kD;
#pragma unroll
  for (int q = 0; q < 4; ++q) {
    int j = tid * 4 + q;
    xo[j] = (e[q] - mu) * rstd * g2[j] + bb2[j] + xr[j];
  }
}

// ---------------------------------------------------------------------------
// Attention: one block per (h, b). K kept in LDS (bf16, padded stride 136),
// softmax in registers via half-wave shuffles, attn restaged via same LDS,
// V B-fragments streamed from global.
// ---------------------------------------------------------------------------
__global__ __launch_bounds__(256) void attention(const float* __restrict__ qkv,
                                                 float* __restrict__ o) {
  __shared__ __bf16 sm[128 * 136];
  int h = blockIdx.x, b = blockIdx.y;
  int tid = threadIdx.x, lane = tid & 31, wv = tid >> 5;
  const float scale = 0.08838834764831845f;  // 1/sqrt(128)
  const float* base = qkv + (size_t)b * kT * (3 * kD);

  // stage K[b,:,h*hd : h*hd+128] -> LDS bf16
  for (int i = tid; i < 128 * 128; i += 256) {
    int r = i >> 7, c = i & 127;
    sm[r * 136 + c] = (__bf16)base[(size_t)r * 3072 + kD + h * kHD + c];
  }
  __syncthreads();

  int t0 = wv * 16;  // 8 waves * 16 query rows = 128

  // Q A-fragments (pre-scaled), K = 4 chunks of 32
  v16bf aq[4];
  {
    int Mr = lane & 15, half = lane >> 4;
    const float* qrow = base + (size_t)(t0 + Mr) * 3072 + h * kHD;
#pragma unroll
    for (int kk = 0; kk < 4; ++kk) {
      v16bf a;
#pragma unroll
      for (int j = 0; j < 8; ++j) {
        int k = kk * 32 + 16 * half + 2 * j;
        a[2 * j]     = (__bf16)(qrow[k] * scale);
        a[2 * j + 1] = (__bf16)(qrow[k + 1] * scale);
      }
      aq[kk] = a;
    }
  }

  // scores strip [16 x 128]
  v8f sc[8];
#pragma unroll
  for (int n = 0; n < 8; ++n) {
    v8f c = zero8();
#pragma unroll
    for (int kk = 0; kk < 4; ++kk) {
      v16bf bf = load_b_frag(sm + kk * 32, 136, n * 16, lane);
      c = wmma_bf16(aq[kk], bf, c);
    }
    sc[n] = c;
  }

  // row softmax (row r lives in one 16-lane half; xor masks 1..8 stay inside)
#pragma unroll
  for (int r = 0; r < 8; ++r) {
    float mx = -3.0e38f;
#pragma unroll
    for (int n = 0; n < 8; ++n) mx = fmaxf(mx, sc[n][r]);
#pragma unroll
    for (int m = 8; m >= 1; m >>= 1) mx = fmaxf(mx, __shfl_xor(mx, m, 32));
    float s = 0.f;
#pragma unroll
    for (int n = 0; n < 8; ++n) {
      float e = __expf(sc[n][r] - mx);
      sc[n][r] = e;
      s += e;
    }
#pragma unroll
    for (int m = 8; m >= 1; m >>= 1) s += __shfl_xor(s, m, 32);
    float inv = 1.f / s;
#pragma unroll
    for (int n = 0; n < 8; ++n) sc[n][r] *= inv;
  }

  __syncthreads();  // everyone done reading K from LDS
  {                 // restage attn (C-layout regs -> row-major LDS) for A-frags
    int half = lane >> 4, c16 = lane & 15;
#pragma unroll
    for (int n = 0; n < 8; ++n)
#pragma unroll
      for (int r = 0; r < 8; ++r)
        sm[(t0 + r + 8 * half) * 136 + n * 16 + c16] = (__bf16)sc[n][r];
  }
  __syncthreads();

  // O strip = attn[16x128] @ V[128x128]; V B-frags from global
  int half = lane >> 4, c16 = lane & 15;
#pragma unroll
  for (int d = 0; d < 8; ++d) {
    v8f c = zero8();
#pragma unroll
    for (int kk = 0; kk < 4; ++kk) {
      v16bf a = load_a_frag(sm + kk * 32, 136, t0, lane);
      v16bf bf;
#pragma unroll
      for (int j = 0; j < 8; ++j) {
        int k = kk * 32 + 16 * half + 2 * j;
        const float* v0 =
            base + (size_t)k * 3072 + 2 * kD + h * kHD + d * 16 + c16;
        bf[2 * j]     = (__bf16)v0[0];
        bf[2 * j + 1] = (__bf16)v0[3072];
      }
      c = wmma_bf16(a, bf, c);
    }
#pragma unroll
    for (int r = 0; r < 8; ++r)
      o[(size_t)(b * kT + t0 + r + 8 * half) * kD + h * kHD + d * 16 + c16] =
          c[r];
  }
}

// ---------------------------------------------------------------------------
// raw = LN(xp + attn_out); in-place safe on xp.
// ---------------------------------------------------------------------------
__global__ __launch_bounds__(256) void resid_ln(const float* __restrict__ xp,
                                                const float* __restrict__ ao,
                                                const float* __restrict__ g,
                                                const float* __restrict__ bb,
                                                float* __restrict__ raw) {
  __shared__ float red[256], red2[256];
  size_t tok = blockIdx.x;
  int tid = threadIdx.x;
  const float* xr = xp + tok * kD;
  const float* ar = ao + tok * kD;
  float v[4];
  float s = 0.f, s2 = 0.f;
#pragma unroll
  for (int q = 0; q < 4; ++q) {
    int j = tid * 4 + q;
    v[q] = xr[j] + ar[j];
    s += v[q];
    s2 += v[q] * v[q];
  }
  red[tid] = s;
  red2[tid] = s2;
  __syncthreads();
  for (int off = 128; off >= 1; off >>= 1) {
    if (tid < off) {
      red[tid] += red[tid + off];
      red2[tid] += red2[tid + off];
    }
    __syncthreads();
  }
  float mu = red[0] / 1024.f;
  float var = red2[0] / 1024.f - mu * mu;
  float rstd = rsqrtf(var + 1e-5f);
  float* rr = raw + tok * kD;
#pragma unroll
  for (int q = 0; q < 4; ++q) {
    int j = tid * 4 + q;
    rr[j] = (v[q] - mu) * rstd * g[j] + bb[j];
  }
}

// ---------------------------------------------------------------------------
// One LSTM time step: gates = xg[:,t,:] + h@Whh^T; update h,c; y[t]=h.
// Block handles 16 gate columns (j0) across all 4 gates; 4 waves = 4 gates,
// each wave: 4 WMMA row-tiles of [64 x 16].
// ---------------------------------------------------------------------------
__global__ __launch_bounds__(128) void lstm_step(
    const float* __restrict__ xg,   // [B*T, 4096] row = b*T + t
    const float* __restrict__ Whh,  // [4096, 1024]
    const float* __restrict__ hin, const float* __restrict__ cin,
    float* __restrict__ hout, float* __restrict__ cout,
    float* __restrict__ yout,  // [B*T, 1024]
    int t) {
  __shared__ __bf16 hs[64 * 40];
  __shared__ __bf16 wsm[4 * 16 * 40];
  __shared__ float gsm[4 * 64 * 16];
  int j0 = blockIdx.x * 16;
  int tid = threadIdx.x, lane = tid & 31, wv = tid >> 5;  // wv = gate 0..3

  v8f acc[4];
#pragma unroll
  for (int m = 0; m < 4; ++m) acc[m] = zero8();

  for (int k0 = 0; k0 < kD; k0 += 32) {
    for (int i = tid; i < 64 * 32; i += 128) {
      int r = i >> 5, c = i & 31;
      hs[r * 40 + c] = (__bf16)hin[(size_t)r * kD + k0 + c];
    }
    for (int i = tid; i < 4 * 16 * 32; i += 128) {
      int g = i >> 9, rem = i & 511;
      int r = rem >> 5, c = rem & 31;
      wsm[g * 640 + r * 40 + c] =
          (__bf16)Whh[(size_t)(g * kD + j0 + r) * kD + k0 + c];
    }
    __syncthreads();
    v16bf bf = load_b_frag(wsm + wv * 640, 40, 0, lane);
#pragma unroll
    for (int m = 0; m < 4; ++m) {
      v16bf a = load_a_frag(hs, 40, m * 16, lane);
      acc[m] = wmma_bf16(a, bf, acc[m]);
    }
    __syncthreads();
  }

  int half = lane >> 4, c16 = lane & 15;
#pragma unroll
  for (int m = 0; m < 4; ++m)
#pragma unroll
    for (int r = 0; r < 8; ++r) {
      int row = m * 16 + r + 8 * half;  // batch index
      float v = acc[m][r] +
                xg[((size_t)row * kT + t) * (4 * kD) + wv * kD + j0 + c16];
      gsm[wv * 1024 + row * 16 + c16] = v;
    }
  __syncthreads();

  for (int i = tid; i < 1024; i += 128) {
    int b = i >> 4, cc = i & 15;
    float iv = gsm[0 * 1024 + i];
    float fv = gsm[1 * 1024 + i];
    float gv = gsm[2 * 1024 + i];
    float ov = gsm[3 * 1024 + i];
    float cp = cin[(size_t)b * kD + j0 + cc];
    float cn = sigf(fv) * cp + sigf(iv) * tanhf(gv);
    float hn = sigf(ov) * tanhf(cn);
    cout[(size_t)b * kD + j0 + cc] = cn;
    hout[(size_t)b * kD + j0 + cc] = hn;
    yout[((size_t)b * kT + t) * kD + j0 + cc] = hn;
  }
}

// ---------------------------------------------------------------------------
// small elementwise / reduction kernels
// ---------------------------------------------------------------------------
__global__ void row_mean(const float* __restrict__ raw,
                         float* __restrict__ rmean) {
  int b = blockIdx.x;
  int d0 = threadIdx.x * 4;
  float a0 = 0, a1 = 0, a2 = 0, a3 = 0;
  for (int t = 0; t < kT; ++t) {
    const float* p = raw + ((size_t)(b * kT + t)) * kD + d0;
    a0 += p[0];
    a1 += p[1];
    a2 += p[2];
    a3 += p[3];
  }
  float* q = rmean + (size_t)b * kD + d0;
  q[0] = a0 * (1.f / kT);
  q[1] = a1 * (1.f / kT);
  q[2] = a2 * (1.f / kT);
  q[3] = a3 * (1.f / kT);
}

__global__ void build_cat(const float* __restrict__ pooled,
                          const float* __restrict__ rmean,
                          float* __restrict__ cat) {
  int i = blockIdx.x * 256 + threadIdx.x;  // 64*2048
  int b = i >> 11, c = i & 2047;
  cat[i] = (c < kD) ? pooled[(size_t)b * kD + c]
                    : rmean[(size_t)b * kD + (c - kD)];
}

__global__ void fuse_gate(const float* __restrict__ g,
                          const float* __restrict__ pooled,
                          const float* __restrict__ rmean,
                          float* __restrict__ fused) {
  int i = blockIdx.x * 256 + threadIdx.x;  // 65536
  float gg = g[i];
  fused[i] = gg * pooled[i] + (1.f - gg) * rmean[i];
}

__global__ __launch_bounds__(256) void head_kernel(
    const float* __restrict__ fused, const float* __restrict__ w1,
    const float* __restrict__ b1, const float* __restrict__ g,
    const float* __restrict__ bb, const float* __restrict__ w2,
    const float* __restrict__ b2, float* __restrict__ logits) {
  __shared__ float hv[512];
  __shared__ float red[256], red2[256];
  int b = blockIdx.x, tid = threadIdx.x;
  const float* f = fused + (size_t)b * kD;
  float loc[2];
  float s = 0.f, s2 = 0.f;
#pragma unroll
  for (int q = 0; q < 2; ++q) {
    int j = tid * 2 + q;
    float acc = b1[j];
    const float* wr = w1 + (size_t)j * kD;
    for (int k = 0; k < kD; ++k) acc += f[k] * wr[k];
    loc[q] = acc;
    s += acc;
    s2 += acc * acc;
  }
  red[tid] = s;
  red2[tid] = s2;
  __syncthreads();
  for (int off = 128; off >= 1; off >>= 1) {
    if (tid < off) {
      red[tid] += red[tid + off];
      red2[tid] += red2[tid + off];
    }
    __syncthreads();
  }
  float mu = red[0] / 512.f;
  float var = red2[0] / 512.f - mu * mu;
  float rstd = rsqrtf(var + 1e-5f);
#pragma unroll
  for (int q = 0; q < 2; ++q) {
    int j = tid * 2 + q;
    hv[j] = fmaxf((loc[q] - mu) * rstd * g[j] + bb[j], 0.f);
  }
  __syncthreads();
  for (int sI = 0; sI < kS; ++sI) {
    float p = 0.f;
    for (int k = tid; k < 512; k += 256) p += hv[k] * w2[(size_t)sI * 512 + k];
    red[tid] = p;
    __syncthreads();
    for (int off = 128; off >= 1; off >>= 1) {
      if (tid < off) red[tid] += red[tid + off];
      __syncthreads();
    }
    if (tid == 0) logits[b * kS + sI] = red[0] + b2[sI];
    __syncthreads();
  }
}

// ---------------------------------------------------------------------------
// memory bank update
// ---------------------------------------------------------------------------
__global__ void fill_f32(float* p, float v, int n) {
  int i = blockIdx.x * 256 + threadIdx.x;
  if (i < n) p[i] = v;
}
__global__ void fill_i32(int* p, int v, int n) {
  int i = blockIdx.x * 256 + threadIdx.x;
  if (i < n) p[i] = v;
}

__global__ void seg_last(const int* __restrict__ class_ids,
                         const int* __restrict__ stage, int* __restrict__ last) {
  int i = blockIdx.x * 256 + threadIdx.x;  // 8192
  int b = i >> 7;
  int pair = class_ids[b] * kS + stage[i];
  atomicMax(&last[pair], i);
}

__global__ __launch_bounds__(256) void mem_update(
    const float* __restrict__ memf, const int* __restrict__ cnts,
    const int* __restrict__ last, const float* __restrict__ raw,
    float* __restrict__ outm) {
  int m = blockIdx.x;     // 0..M-1
  int pair = blockIdx.y;  // 0..C*S-1
  int tid = threadIdx.x;
  int cnt = cnts[pair];
  int ls = last[pair];
  bool present = ls >= 0;
  bool full = cnt >= kM;
  const float* src;
  if (present && full)
    src = (m < kM - 1) ? memf + ((size_t)pair * kM + m + 1) * kD
                       : raw + (size_t)ls * kD;
  else if (present && m == cnt)
    src = raw + (size_t)ls * kD;
  else
    src = memf + ((size_t)pair * kM + m) * kD;
  float* dst = outm + ((size_t)pair * kM + m) * kD;
  for (int c = tid; c < kD; c += 256) dst[c] = src[c];
}

__global__ void cnt_update(const int* __restrict__ cnts,
                           const int* __restrict__ last, int* __restrict__ outc) {
  int i = blockIdx.x * 256 + threadIdx.x;
  if (i >= kC * kS) return;
  int c = cnts[i];
  outc[i] = c + ((last[i] >= 0 && c < kM) ? 1 : 0);
}

// ---------------------------------------------------------------------------
// host launcher
// ---------------------------------------------------------------------------
extern "C" void kernel_launch(void* const* d_in, const int* in_sizes, int n_in,
                              void* d_out, int out_size, void* d_ws,
                              size_t ws_size, hipStream_t stream) {
  (void)in_sizes; (void)n_in; (void)out_size; (void)ws_size;

  const float* x       = (const float*)d_in[0];
  const int*   stage   = (const int*)d_in[1];
  const int*   cls     = (const int*)d_in[2];
  const float* enc_w1  = (const float*)d_in[3];
  const float* enc_b1  = (const float*)d_in[4];
  const float* enc_g1  = (const float*)d_in[5];
  const float* enc_bb1 = (const float*)d_in[6];
  const float* enc_w2  = (const float*)d_in[7];
  const float* enc_b2  = (const float*)d_in[8];
  const float* enc_g2  = (const float*)d_in[9];
  const float* enc_bb2 = (const float*)d_in[10];
  const float* qkv_w   = (const float*)d_in[11];
  const float* qkv_b   = (const float*)d_in[12];
  const float* out_w   = (const float*)d_in[13];
  const float* out_b   = (const float*)d_in[14];
  const float* norm_g  = (const float*)d_in[15];
  const float* norm_b  = (const float*)d_in[16];
  const float* Wih0    = (const float*)d_in[17];
  const float* Whh0    = (const float*)d_in[18];
  const float* bih0    = (const float*)d_in[19];
  const float* bhh0    = (const float*)d_in[20];
  const float* Wih1    = (const float*)d_in[21];
  const float* Whh1    = (const float*)d_in[22];
  const float* bih1    = (const float*)d_in[23];
  const float* bhh1    = (const float*)d_in[24];
  const float* head_w1 = (const float*)d_in[25];
  const float* head_b1 = (const float*)d_in[26];
  const float* head_g  = (const float*)d_in[27];
  const float* head_bb = (const float*)d_in[28];
  const float* head_w2 = (const float*)d_in[29];
  const float* head_b2 = (const float*)d_in[30];
  const float* gate_w1 = (const float*)d_in[31];
  const float* gate_b1 = (const float*)d_in[32];
  const float* gate_w2 = (const float*)d_in[33];
  const float* gate_b2 = (const float*)d_in[34];
  const float* memf    = (const float*)d_in[35];
  const int*   memcnt  = (const int*)d_in[36];

  float* out      = (float*)d_out;
  float* fused_o  = out;                       // 65536
  float* raw_o    = out + 65536;               // 8388608  (also hosts xp)
  float* logits_o = out + 8454144;             // 320
  float* mem_o    = out + 8454464;             // 51200000
  int*   cnt_o    = (int*)(out + 59654464);    // 5000

  float* ws = (float*)d_ws;
  // workspace layout (floats), aliased where lifetimes allow
  float* WS_QKV = ws + 0;          // [8192,3072]   25165824
  float* WS_O   = ws + 25165824;   // [8192,1024]    8388608
  float* WS_AO  = ws + 33554432;   // [8192,1024]    8388608
  float* WS_XG  = ws + 0;          // [8192,4096]   33554432  (reuses QKV+O)
  float* WS_Y0  = ws + 33554432;   // [8192,1024]             (reuses AO)
  const size_t SB = 41943040;
  float* H0     = ws + SB;             // [64,1024]
  float* H1     = ws + SB + 65536;
  float* C0     = ws + SB + 131072;
  float* C1     = ws + SB + 196608;
  float* RMEAN  = ws + SB + 262144;
  float* CAT    = ws + SB + 327680;    // [64,2048]
  float* G1B    = ws + SB + 458752;
  float* GGB    = ws + SB + 524288;
  int*   LAST   = (int*)(ws + SB + 589824);  // [5000]

  // 1) stage encoder + residual add -> xp (stored in raw slot)
  encode_add<<<kBT, 256, 0, stream>>>(x, stage, enc_w1, enc_b1, enc_g1, enc_bb1,
                                      enc_w2, enc_b2, enc_g2, enc_bb2, raw_o);
  // 2) QKV projection
  gemm_bias_act<<<dim3(kBT / 64, 3072 / 64), 128, 0, stream>>>(
      raw_o, qkv_w, qkv_b, nullptr, WS_QKV, kBT, kD, 3072, 0);
  // 3) attention
  attention<<<dim3(kH, kB), 256, 0, stream>>>(WS_QKV, WS_O);
  // 4) output projection
  gemm_bias_act<<<dim3(kBT / 64, kD / 64), 128, 0, stream>>>(
      WS_O, out_w, out_b, nullptr, WS_AO, kBT, kD, kD, 0);
  // 5) residual + LN -> raw (in-place over xp)
  resid_ln<<<kBT, 256, 0, stream>>>(raw_o, WS_AO, norm_g, norm_b, raw_o);

  // 6) LSTM layer 0: pre-gates = raw @ Wih0^T + (bih0+bhh0)
  gemm_bias_act<<<dim3(kBT / 64, 4096 / 64), 128, 0, stream>>>(
      raw_o, Wih0, bih0, bhh0, WS_XG, kBT, kD, 4096, 0);
  fill_f32<<<(4 * 65536 + 255) / 256, 256, 0, stream>>>(H0, 0.f, 4 * 65536);
  for (int t = 0; t < kT; ++t) {
    float* hin  = (t & 1) ? H1 : H0;
    float* hout = (t & 1) ? H0 : H1;
    float* cin  = (t & 1) ? C1 : C0;
    float* cout = (t & 1) ? C0 : C1;
    lstm_step<<<64, 128, 0, stream>>>(WS_XG, Whh0, hin, cin, hout, cout, WS_Y0,
                                      t);
  }
  // 7) LSTM layer 1
  gemm_bias_act<<<dim3(kBT / 64, 4096 / 64), 128, 0, stream>>>(
      WS_Y0, Wih1, bih1, bhh1, WS_XG, kBT, kD, 4096, 0);
  fill_f32<<<(4 * 65536 + 255) / 256, 256, 0, stream>>>(H0, 0.f, 4 * 65536);
  for (int t = 0; t < kT; ++t) {
    float* hin  = (t & 1) ? H1 : H0;
    float* hout = (t & 1) ? H0 : H1;
    float* cin  = (t & 1) ? C1 : C0;
    float* cout = (t & 1) ? C0 : C1;
    lstm_step<<<64, 128, 0, stream>>>(WS_XG, Whh1, hin, cin, hout, cout, WS_Y0,
                                      t);
  }
  float* POOLED = H0;  // t=127 (odd) writes into H0

  // 8) gated fusion
  row_mean<<<kB, 256, 0, stream>>>(raw_o, RMEAN);
  build_cat<<<(kB * 2048) / 256, 256, 0, stream>>>(POOLED, RMEAN, CAT);
  gemm_bias_act<<<dim3(1, kD / 64), 128, 0, stream>>>(CAT, gate_w1, gate_b1,
                                                      nullptr, G1B, kB, 2 * kD,
                                                      kD, 1);
  gemm_bias_act<<<dim3(1, kD / 64), 128, 0, stream>>>(G1B, gate_w2, gate_b2,
                                                      nullptr, GGB, kB, kD, kD,
                                                      2);
  fuse_gate<<<(kB * kD) / 256, 256, 0, stream>>>(GGB, POOLED, RMEAN, fused_o);

  // 9) stage head
  head_kernel<<<kB, 256, 0, stream>>>(fused_o, head_w1, head_b1, head_g,
                                      head_bb, head_w2, head_b2, logits_o);

  // 10) memory bank update
  fill_i32<<<(kC * kS + 255) / 256, 256, 0, stream>>>(LAST, -1, kC * kS);
  seg_last<<<kBT / 256, 256, 0, stream>>>(cls, stage, LAST);
  mem_update<<<dim3(kM, kC * kS), 256, 0, stream>>>(memf, memcnt, LAST, raw_o,
                                                    mem_o);
  cnt_update<<<(kC * kS + 255) / 256, 256, 0, stream>>>(memcnt, LAST, cnt_o);
}
